// _SingleHeadCausalAttention_15058155340486
// MI455X (gfx1250) — matrix operand order repaired and checked by reading, not compiled, not run on hardware
//
#include <hip/hip_runtime.h>
#include <hip/hip_bf16.h>

// ---------- types ----------
typedef __bf16 bf16_t;
typedef __attribute__((ext_vector_type(16))) __bf16 v16bf;
typedef __attribute__((ext_vector_type(8)))  __bf16 v8bf;
typedef __attribute__((ext_vector_type(8)))  float  v8f;

#define B_  4
#define T_  2048
#define E_  1024
#define H_  64

// Concatenate two 16-byte halves into one 32-byte fragment (register concat).
__device__ __forceinline__ v16bf concat8(v8bf lo, v8bf hi) {
  struct P { v8bf a, b; } p{lo, hi};
  return __builtin_bit_cast(v16bf, p);
}

// Native RNE fp32 -> bf16 (v_cvt_pk_bf16_f32 when vectorized)
__device__ __forceinline__ bf16_t f2bf(float f) { return (bf16_t)f; }

__device__ __forceinline__ v8f do_wmma(v16bf a, v16bf b, v8f c) {
  return __builtin_amdgcn_wmma_f32_16x16x32_bf16(
      /*neg_a=*/false, a, /*neg_b=*/false, b,
      /*c_mod=*/(short)0, c, /*reuse_a=*/false, /*reuse_b=*/false);
}

// A-fragment (16x32 bf16): lane half selects K {0..15} / {16..31}
__device__ __forceinline__ v16bf load_a_bf16(const bf16_t* prow, int half) {
  const bf16_t* p = prow + (half ? 16 : 0);
  return concat8(*(const v8bf*)(p), *(const v8bf*)(p + 8));
}

// B-fragment (32x16 bf16): lane holds 16 contiguous K at (half?16:0)
__device__ __forceinline__ v16bf load_b_bf16(const bf16_t* prow, int half) {
  const bf16_t* p = prow + (half ? 16 : 0);
  return concat8(*(const v8bf*)(p), *(const v8bf*)(p + 8));
}

// ============================================================
// Prep: bulk fp32 -> bf16 conversion (bandwidth-bound, one shot).
// ============================================================
__global__ __launch_bounds__(256) void cvt_bf16_kernel(
    const float* __restrict__ src, bf16_t* __restrict__ dst, int n8)
{
  int i = blockIdx.x * 256 + threadIdx.x;
  if (i < n8) {
    v8f v = *((const v8f*)src + i);
    v8bf o;
#pragma unroll
    for (int j = 0; j < 8; ++j) o[j] = f2bf(v[j]);
    *((v8bf*)dst + i) = o;
  }
}

// ============================================================
// QKV projection (pure bf16 WMMA), compile-time specialized:
//   M=0: Q -> Qs[b][t][h] * 0.125   M=1: K -> Kk[b][t][h]
//   M=2: V -> Vt[b][h][t] (transposed)
// Software-pipelined: chunk i+1's loads are issued BEFORE chunk i's
// WMMAs, so the matrix ops retire on partial loadcnt waits while the
// next fragment set streams in.
// ============================================================
template <int M>
__device__ __forceinline__ void proj_pass(
    const bf16_t* __restrict__ xrow, const bf16_t* __restrict__ W,
    const float* __restrict__ bias, bf16_t* __restrict__ dstQK,
    bf16_t* __restrict__ dstVt, int b, int t0, int half, int lr)
{
  v8f acc[4] = {v8f{}, v8f{}, v8f{}, v8f{}};
  const bf16_t* w0 = W + (size_t)(0 * 16 + lr) * E_;
  const bf16_t* w1 = W + (size_t)(1 * 16 + lr) * E_;
  const bf16_t* w2 = W + (size_t)(2 * 16 + lr) * E_;
  const bf16_t* w3 = W + (size_t)(3 * 16 + lr) * E_;

  // pipeline prologue: chunk 0 fragments in flight
  v16bf a  = load_a_bf16(xrow, half);
  v16bf b0 = load_b_bf16(w0, half);
  v16bf b1 = load_b_bf16(w1, half);
  v16bf b2 = load_b_bf16(w2, half);
  v16bf b3 = load_b_bf16(w3, half);

#pragma unroll 2
  for (int ec = 32; ec < E_; ec += 32) {
    // issue next chunk's loads first (independent register set)
    v16bf an  = load_a_bf16(xrow + ec, half);
    v16bf bn0 = load_b_bf16(w0 + ec, half);
    v16bf bn1 = load_b_bf16(w1 + ec, half);
    v16bf bn2 = load_b_bf16(w2 + ec, half);
    v16bf bn3 = load_b_bf16(w3 + ec, half);
    // consume previous chunk (waits only for its own loads)
    acc[0] = do_wmma(a, b0, acc[0]);
    acc[1] = do_wmma(a, b1, acc[1]);
    acc[2] = do_wmma(a, b2, acc[2]);
    acc[3] = do_wmma(a, b3, acc[3]);
    a = an; b0 = bn0; b1 = bn1; b2 = bn2; b3 = bn3;  // rotate (renamed away)
  }
  // pipeline epilogue
  acc[0] = do_wmma(a, b0, acc[0]);
  acc[1] = do_wmma(a, b1, acc[1]);
  acc[2] = do_wmma(a, b2, acc[2]);
  acc[3] = do_wmma(a, b3, acc[3]);

#pragma unroll
  for (int f = 0; f < 4; ++f) {
    const int h = f * 16 + lr;              // C-frag: N = lane%16
    const float bb = bias[h];
    if (M == 2) {
      // transposed store: rows (M=r+8*half) contiguous in t -> one b128 store
      v8bf pk;
#pragma unroll
      for (int r = 0; r < 8; ++r) pk[r] = f2bf(acc[f][r] + bb);
      *(v8bf*)(dstVt + ((size_t)b * H_ + h) * T_ + t0 + half * 8) = pk;
    } else {
#pragma unroll
      for (int r = 0; r < 8; ++r) {
        const int t = t0 + r + half * 8;    // C-frag: M = r + 8*half
        float val = acc[f][r] + bb;
        if (M == 0) val *= 0.125f;          // fold 1/sqrt(64) into Q
        dstQK[((size_t)b * T_ + t) * H_ + h] = f2bf(val);
      }
    }
  }
}

__global__ __launch_bounds__(128) void qkv_proj_kernel(
    const bf16_t* __restrict__ xbf, const bf16_t* __restrict__ Wbf,
    const float* __restrict__ bq, const float* __restrict__ bk,
    const float* __restrict__ bv,
    bf16_t* __restrict__ Qs, bf16_t* __restrict__ Kk, bf16_t* __restrict__ Vt)
{
  const int lane = threadIdx.x & 31;
  const int wave = threadIdx.x >> 5;
  const int tile = blockIdx.x * 4 + wave;   // 0..511
  const int b    = tile >> 7;
  const int it   = tile & 127;
  const int t0   = it * 16;
  const int half = lane >> 4;
  const int lr   = lane & 15;

  const bf16_t* xrow = xbf + ((size_t)b * T_ + (t0 + lr)) * E_;
  const bf16_t* Wq = Wbf;                   // [64][1024] each, bf16
  const bf16_t* Wk = Wbf + (size_t)H_ * E_;
  const bf16_t* Wv = Wbf + (size_t)2 * H_ * E_;

  proj_pass<0>(xrow, Wq, bq, Qs, nullptr, b, t0, half, lr);
  proj_pass<1>(xrow, Wk, bk, Kk, nullptr, b, t0, half, lr);
  proj_pass<2>(xrow, Wv, bv, nullptr, Vt, b, t0, half, lr);
}

// ============================================================
// Flash-style causal attention: 4 waves/block, one 16-query tile
// per wave, streaming 32-key chunks up to the diagonal.
// ============================================================
__global__ __launch_bounds__(128) void attn_kernel(
    const bf16_t* __restrict__ Qs, const bf16_t* __restrict__ Kk,
    const bf16_t* __restrict__ Vt, float* __restrict__ out)
{
  __shared__ bf16_t lds_p[4][16 * 32];       // per-wave P tile (bf16)

  const int lane = threadIdx.x & 31;
  const int wave = threadIdx.x >> 5;
  const int tile = blockIdx.x * 4 + wave;
  const int b    = tile >> 7;
  const int it   = tile & 127;
  const int t0   = it * 16;
  const int half = lane >> 4;
  const int lr   = lane & 15;

  const bf16_t* Qb = Qs + (size_t)b * T_ * H_;
  const bf16_t* Kb = Kk + (size_t)b * T_ * H_;
  const bf16_t* Vb = Vt + (size_t)b * H_ * T_;

  // Hoisted Q A-fragments (K-dim = 64 -> two 16x32 fragments)
  const bf16_t* qrow = Qb + (size_t)(t0 + lr) * H_;
  v16bf qA0 = load_a_bf16(qrow, half);
  v16bf qA1 = load_a_bf16(qrow + 32, half);

  v8f O[4] = {v8f{}, v8f{}, v8f{}, v8f{}};
  float mrow[8], lrow[8];
#pragma unroll
  for (int r = 0; r < 8; ++r) { mrow[r] = -__builtin_inff(); lrow[r] = 0.0f; }

  bf16_t* myP = lds_p[wave];
  const int nch = (it >> 1) + 1;

#pragma unroll 1
  for (int c = 0; c < nch; ++c) {
    const int sb = c * 32;

    if (c + 1 < nch) {                       // prefetch next chunk's K rows
      __builtin_prefetch(Kb + (size_t)(sb + 32 + lr) * H_, 0, 3);
      __builtin_prefetch(Kb + (size_t)(sb + 48 + lr) * H_, 0, 3);
    }

    // ---- S = Q * K^T : 16x32 scores as two 16x16 C-fragments ----
    v16bf b00 = load_b_bf16(Kb + (size_t)(sb + lr) * H_, half);
    v16bf b10 = load_b_bf16(Kb + (size_t)(sb + 16 + lr) * H_, half);
    v16bf b01 = load_b_bf16(Kb + (size_t)(sb + lr) * H_ + 32, half);
    v16bf b11 = load_b_bf16(Kb + (size_t)(sb + 16 + lr) * H_ + 32, half);
    v8f S0 = v8f{}, S1 = v8f{};
    S0 = do_wmma(qA0, b00, S0);
    S1 = do_wmma(qA0, b10, S1);
    S0 = do_wmma(qA1, b01, S0);
    S1 = do_wmma(qA1, b11, S1);

    // ---- issue V B-fragment loads EARLY: independent of softmax/LDS,
    //      they overlap the ~200 VALU ops of the softmax section ----
    v16bf vB0 = load_b_bf16(Vb + (size_t)(0 * 16 + lr) * T_ + sb, half);
    v16bf vB1 = load_b_bf16(Vb + (size_t)(1 * 16 + lr) * T_ + sb, half);
    v16bf vB2 = load_b_bf16(Vb + (size_t)(2 * 16 + lr) * T_ + sb, half);
    v16bf vB3 = load_b_bf16(Vb + (size_t)(3 * 16 + lr) * T_ + sb, half);

    // ---- causal mask + online softmax (fp32) ----
    const int s0col = sb + lr;               // C-frag: N = lane%16
    const int s1col = sb + 16 + lr;
#pragma unroll
    for (int r = 0; r < 8; ++r) {
      const int q = t0 + r + half * 8;       // C-frag: M = r + 8*half
      float v0 = (s0col <= q) ? S0[r] : -__builtin_inff();
      float v1 = (s1col <= q) ? S1[r] : -__builtin_inff();
      float cm = fmaxf(v0, v1);
      cm = fmaxf(cm, __shfl_xor(cm, 1));
      cm = fmaxf(cm, __shfl_xor(cm, 2));
      cm = fmaxf(cm, __shfl_xor(cm, 4));
      cm = fmaxf(cm, __shfl_xor(cm, 8));     // row reduce within 16-lane half
      const float mn    = fmaxf(mrow[r], cm);
      const float alpha = __expf(mrow[r] - mn);
      mrow[r] = mn;
      const float e0 = __expf(v0 - mn);
      const float e1 = __expf(v1 - mn);
      float rs = e0 + e1;
      rs += __shfl_xor(rs, 1); rs += __shfl_xor(rs, 2);
      rs += __shfl_xor(rs, 4); rs += __shfl_xor(rs, 8);
      lrow[r] = lrow[r] * alpha + rs;
      O[0][r] *= alpha; O[1][r] *= alpha; O[2][r] *= alpha; O[3][r] *= alpha;
      const int row = r + half * 8;          // stage P row-major [m][s]
      myP[row * 32 + lr]      = f2bf(e0);
      myP[row * 32 + 16 + lr] = f2bf(e1);
    }

    asm volatile("" ::: "memory");           // keep DS store->load program order

    // ---- reload P as A-fragment (DS pipeline is in-order per wave) ----
    v16bf pA = load_a_bf16(myP + lr * 32, half);

    // ---- O += P * V ----
    O[0] = do_wmma(pA, vB0, O[0]);
    O[1] = do_wmma(pA, vB1, O[1]);
    O[2] = do_wmma(pA, vB2, O[2]);
    O[3] = do_wmma(pA, vB3, O[3]);
  }

  // ---- epilogue: normalize and store fp32 ----
  float inv[8];
#pragma unroll
  for (int r = 0; r < 8; ++r) inv[r] = 1.0f / lrow[r];
#pragma unroll
  for (int f = 0; f < 4; ++f) {
    const int h = f * 16 + lr;
#pragma unroll
    for (int r = 0; r < 8; ++r) {
      const int t = t0 + r + half * 8;
      out[((size_t)b * T_ + t) * H_ + h] = O[f][r] * inv[r];
    }
  }
}

// ============================================================
extern "C" void kernel_launch(void* const* d_in, const int* in_sizes, int n_in,
                              void* d_out, int out_size, void* d_ws, size_t ws_size,
                              hipStream_t stream) {
  const float* x  = (const float*)d_in[0];
  const float* Wq = (const float*)d_in[1];
  const float* bq = (const float*)d_in[2];
  const float* Wk = (const float*)d_in[3];
  const float* bk = (const float*)d_in[4];
  const float* Wv = (const float*)d_in[5];
  const float* bv = (const float*)d_in[6];
  float* out = (float*)d_out;

  // workspace layout (bf16):
  //   Qs [B,T,H] | Kk [B,T,H] | Vt [B,H,T] | xbf [B,T,E] | Wbf [3,H,E]
  bf16_t* Qs  = (bf16_t*)d_ws;
  bf16_t* Kk  = Qs  + (size_t)B_ * T_ * H_;
  bf16_t* Vt  = Kk  + (size_t)B_ * T_ * H_;
  bf16_t* xbf = Vt  + (size_t)B_ * T_ * H_;
  bf16_t* Wbf = xbf + (size_t)B_ * T_ * E_;

  // one-shot bf16 staging (bandwidth-bound)
  {
    const int n8x = (B_ * T_ * E_) / 8;      // 1,048,576
    cvt_bf16_kernel<<<(n8x + 255) / 256, 256, 0, stream>>>(x, xbf, n8x);
    const int n8w = (H_ * E_) / 8;           // 8,192
    cvt_bf16_kernel<<<(n8w + 255) / 256, 256, 0, stream>>>(Wq, Wbf, n8w);
    cvt_bf16_kernel<<<(n8w + 255) / 256, 256, 0, stream>>>(Wk, Wbf + (size_t)H_ * E_, n8w);
    cvt_bf16_kernel<<<(n8w + 255) / 256, 256, 0, stream>>>(Wv, Wbf + (size_t)2 * H_ * E_, n8w);
  }

  qkv_proj_kernel<<<128, 128, 0, stream>>>(xbf, Wbf, bq, bk, bv, Qs, Kk, Vt);
  attn_kernel<<<128, 128, 0, stream>>>(Qs, Kk, Vt, out);
}
